// BasicUformerLayer_77635828843211
// MI455X (gfx1250) — matrix operand order repaired and checked by reading, not compiled, not run on hardware
//
#include <hip/hip_runtime.h>
#include <hip/hip_bf16.h>
#include <math.h>

typedef __attribute__((ext_vector_type(16))) _Float16 v16h;
typedef __attribute__((ext_vector_type(8)))  _Float16 v8h;
typedef __attribute__((ext_vector_type(8)))  float    v8f;
typedef __attribute__((ext_vector_type(4)))  int      v4i;

#define CDIM   512
#define HEADS  16
#define HD     32
#define WS     8
#define NTOK   64          // tokens per window (8x8)
#define HGRID  128
#define BATCH  2
#define LTOK   (HGRID*HGRID)        // 16384
#define MROWS  (BATCH*LTOK)         // 32768
#define HID    2048
#define NWIN   (HGRID/WS)           // 16 windows per side
#define BW     (BATCH*NWIN*NWIN)    // 512 windows total

#if defined(__gfx1250__) && __has_builtin(__builtin_amdgcn_global_load_async_to_lds_b128) && \
    __has_builtin(__builtin_amdgcn_s_wait_asynccnt)
#define HAS_ASYNC_LDS 1
#else
#define HAS_ASYNC_LDS 0
#endif

static __device__ inline v16h cat8(v8h lo, v8h hi) {
    return __builtin_shufflevector(lo, hi, 0,1,2,3,4,5,6,7,8,9,10,11,12,13,14,15);
}
static __device__ inline v8f wmma16(v16h a, v16h b, v8f c) {
    // D = A(16x32 f16) * B(32x16 f16) + C(16x16 f32)
    return __builtin_amdgcn_wmma_f32_16x16x32_f16(false, a, false, b, (short)0, c, false, false);
}
static __device__ inline v8f zero8() { v8f z = {0.f,0.f,0.f,0.f,0.f,0.f,0.f,0.f}; return z; }

// 16-byte global -> LDS copy: async DMA path on gfx1250 (tracked by ASYNCcnt);
// builtin prototype (from probe): (int4 __device__*, int4 __shared__*, imm, imm)
static __device__ inline void cp16_g2l(const _Float16* g, _Float16* l) {
#if HAS_ASYNC_LDS
    __builtin_amdgcn_global_load_async_to_lds_b128(
        (__attribute__((address_space(1))) v4i*)g,
        (__attribute__((address_space(3))) v4i*)l, 0, 0);
#else
    *(v8h*)l = *(const v8h*)g;
#endif
}
static __device__ inline void cp_fence() {
#if HAS_ASYNC_LDS
    __builtin_amdgcn_s_wait_asynccnt(0);
#endif
}

// ---------------------------------------------------------------------------
// fp32 -> fp16 weight conversion
// ---------------------------------------------------------------------------
__global__ void cvt_f32_f16_kernel(const float* __restrict__ src,
                                   _Float16* __restrict__ dst, int n) {
    int i = blockIdx.x * blockDim.x + threadIdx.x;
    if (i < n) dst[i] = (_Float16)src[i];
}

// ---------------------------------------------------------------------------
// LayerNorm (+ optional shift-roll + window partition), fp32 -> fp16
// One wave per token. blockDim = 128 (4 waves).
// ---------------------------------------------------------------------------
__global__ __launch_bounds__(128)
void ln_kernel(const float* __restrict__ x, const float* __restrict__ g,
               const float* __restrict__ b, _Float16* __restrict__ out,
               int shift, int windowed) {
    const int wave = threadIdx.x >> 5;
    const int lane = threadIdx.x & 31;
    const int tok  = blockIdx.x * 4 + wave;

    const float* xr = x + (size_t)tok * CDIM;
    float v[16];
    float s = 0.f, s2 = 0.f;
    #pragma unroll
    for (int i = 0; i < 16; i++) {
        float t = xr[i * 32 + lane];
        v[i] = t; s += t; s2 += t * t;
    }
    #pragma unroll
    for (int m = 16; m >= 1; m >>= 1) { s += __shfl_xor(s, m, 32); s2 += __shfl_xor(s2, m, 32); }
    const float mu  = s * (1.f / CDIM);
    const float var = s2 * (1.f / CDIM) - mu * mu;
    const float inv = rsqrtf(var + 1e-5f);

    size_t drow;
    if (windowed) {
        int bb = tok >> 14, rem = tok & (LTOK - 1);
        int h = rem >> 7, w = rem & 127;
        int hr = (h - shift) & 127, wr = (w - shift) & 127;
        int win = bb * (NWIN * NWIN) + (hr >> 3) * NWIN + (wr >> 3);
        int n   = (hr & 7) * WS + (wr & 7);
        drow = (size_t)win * NTOK + n;
    } else {
        drow = (size_t)tok;
    }
    _Float16* orow = out + drow * CDIM;
    #pragma unroll
    for (int i = 0; i < 16; i++) {
        int c = i * 32 + lane;
        orow[c] = (_Float16)((v[i] - mu) * inv * g[c] + b[c]);
    }
}

// ---------------------------------------------------------------------------
// Tiled WMMA GEMM:  out(M,N) = A(M,K)f16 * B(K,N)f16 + bias, + epilogue
// block tile 256x128, K-step 32, 256 threads = 8 waves (4x2), wave tile 64x64
// (16 WMMAs per 8 fragment loads per K-step). Double-buffered LDS: one
// barrier per K-step, staging overlapped with the WMMA chain (async-LDS DMA
// when the toolchain exposes it).
// mode 0: f16 store (QKV)        mode 1: GELU + f16 store (fc1)
// mode 2: +resid, f32 store (fc2) mode 3: window-reverse map, +resid, f32 (proj)
// ---------------------------------------------------------------------------
__global__ __launch_bounds__(256)
void gemm_wmma_kernel(const _Float16* __restrict__ A, const _Float16* __restrict__ B,
                      const float* __restrict__ bias, int M, int N, int K,
                      _Float16* __restrict__ outH, float* __restrict__ outF,
                      const float* __restrict__ resid, int mode, int shift) {
    __shared__ _Float16 As[2][256][32 + 8];
    __shared__ _Float16 Bs[2][32][128 + 8];

    const int tid  = threadIdx.x;
    const int lane = tid & 31;
    const int wave = tid >> 5;
    const int wm   = wave >> 1;          // 0..3
    const int wn   = wave & 1;           // 0..1
    const int ml   = lane & 15;
    const int kh   = lane >> 4;

    const int n0 = blockIdx.x * 128;
    const int m0 = blockIdx.y * 256;

    v8f acc[4][4];
    #pragma unroll
    for (int i = 0; i < 4; i++)
        #pragma unroll
        for (int j = 0; j < 4; j++) acc[i][j] = zero8();

    // stage A tile 256x32 (4 chunks/thread) and B tile 32x128 (2 chunks/thread)
    auto stage = [&](int p, int k0) {
        #pragma unroll
        for (int i = 0; i < 4; i++) {
            int ch = tid * 4 + i;
            int r = ch >> 2, cc = ch & 3;
            cp16_g2l(A + (size_t)(m0 + r) * K + k0 + cc * 8, &As[p][r][cc * 8]);
        }
        #pragma unroll
        for (int i = 0; i < 2; i++) {
            int ch = tid * 2 + i;
            int r = ch >> 4, cc = ch & 15;
            cp16_g2l(B + (size_t)(k0 + r) * N + n0 + cc * 8, &Bs[p][r][cc * 8]);
        }
    };

    const int niter = K / 32;
    stage(0, 0);
    cp_fence();
    __syncthreads();

    for (int it = 0; it < niter; it++) {
        const int p = it & 1;
        if (it + 1 < niter) {
            stage(p ^ 1, (it + 1) * 32);
            __builtin_prefetch(A + (size_t)(m0 + tid) * K + (it + 1) * 32, 0, 0);
        }
        // B fragments: lane n = ml, halves: K = kh*8..+7 and 16+kh*8..+7
        v16h bf[4];
        #pragma unroll
        for (int nt = 0; nt < 4; nt++) {
            int c0 = wn * 64 + nt * 16;
            #pragma unroll
            for (int s = 0; s < 8; s++) {
                bf[nt][s]     = Bs[p][kh * 8 + s][c0 + ml];
                bf[nt][8 + s] = Bs[p][16 + kh * 8 + s][c0 + ml];
            }
        }
        // A fragments: lane m = ml, halves: K = kh*8..+7 and 16+kh*8..+7
        #pragma unroll
        for (int mt = 0; mt < 4; mt++) {
            int r = wm * 64 + mt * 16 + ml;
            v16h af = cat8(*(const v8h*)&As[p][r][kh * 8], *(const v8h*)&As[p][r][16 + kh * 8]);
            #pragma unroll
            for (int nt = 0; nt < 4; nt++) acc[mt][nt] = wmma16(af, bf[nt], acc[mt][nt]);
        }
        cp_fence();
        __syncthreads();
    }

    // epilogue — C/D layout: vgpr j, row = j + 8*(lane>>4), col = lane&15
    #pragma unroll
    for (int mt = 0; mt < 4; mt++) {
        #pragma unroll
        for (int nt = 0; nt < 4; nt++) {
            #pragma unroll
            for (int j = 0; j < 8; j++) {
                int row = m0 + wm * 64 + mt * 16 + j + 8 * kh;
                int col = n0 + wn * 64 + nt * 16 + ml;
                float v = acc[mt][nt][j] + bias[col];
                if (mode == 0) {
                    outH[(size_t)row * N + col] = (_Float16)v;
                } else if (mode == 1) {
                    float gl = 0.5f * v * (1.f + erff(v * 0.70710678118654752f));
                    outH[(size_t)row * N + col] = (_Float16)gl;
                } else if (mode == 2) {
                    size_t o = (size_t)row * N + col;
                    outF[o] = v + resid[o];
                } else { // mode 3: windowed row -> image row (undo shift)
                    int win = row >> 6, n = row & 63;
                    int bb = win >> 8, wl = win & 255;
                    int hr = (wl >> 4) * WS + (n >> 3);
                    int wr = (wl & 15) * WS + (n & 7);
                    int h = (hr + shift) & 127, w = (wr + shift) & 127;
                    size_t o = ((size_t)bb * LTOK + h * HGRID + w) * (size_t)N + col;
                    outF[o] = v + resid[o];
                }
            }
        }
    }
}

// ---------------------------------------------------------------------------
// Fused window attention. One wave per (window, head). blockDim = 32.
// S = scale*q.kT + rpb (+ shift mask), softmax, O = P.V   (all WMMA f16)
// ---------------------------------------------------------------------------
static __device__ inline int regionOf(int h) { return (h < HGRID - WS) ? 0 : ((h < HGRID - WS/2) ? 1 : 2); }

__global__ __launch_bounds__(32)
void attn_kernel(const _Float16* __restrict__ qkv, const float* __restrict__ rpb,
                 _Float16* __restrict__ out, int shift) {
    __shared__ _Float16 kT[HD][NTOK + 8];     // k transposed
    __shared__ _Float16 P[NTOK][NTOK + 8];    // softmax probabilities

    const int bid  = blockIdx.x;
    const int win  = bid >> 4;
    const int head = bid & 15;
    const int lane = threadIdx.x;
    const int ml   = lane & 15;
    const int kh   = lane >> 4;
    const int wl   = win & 255;
    const int wh   = wl >> 4, ww = wl & 15;
    const float scale = 0.17677669529663687f;   // 32^-0.5

    const _Float16* qb = qkv + (size_t)win * NTOK * (3 * CDIM) + head * HD;

    // stage kT: lane handles token rows lane and lane+32
    #pragma unroll
    for (int rr = 0; rr < 2; rr++) {
        int n = lane + rr * 32;
        const _Float16* kr = qb + (size_t)n * (3 * CDIM) + CDIM;
        #pragma unroll
        for (int kk = 0; kk < HD; kk++) kT[kk][n] = kr[kk];
    }
    __syncthreads();

    const float* rpbh = rpb + head;

    for (int rh = 0; rh < 2; rh++) {              // two 32-row halves
        // B fragments of kT (lane = key token ml, halves: hd = kh*8..+7 and 16+kh*8..+7)
        v16h bk[4];
        #pragma unroll
        for (int tc = 0; tc < 4; tc++)
            #pragma unroll
            for (int s = 0; s < 8; s++) {
                bk[tc][s]     = kT[kh * 8 + s][tc * 16 + ml];
                bk[tc][8 + s] = kT[16 + kh * 8 + s][tc * 16 + ml];
            }

        v8f S[2][4];
        #pragma unroll
        for (int trl = 0; trl < 2; trl++) {
            int n = (rh * 2 + trl) * 16 + ml;
            const _Float16* qr = qb + (size_t)n * (3 * CDIM);
            v16h qa = cat8(*(const v8h*)(qr + kh * 8), *(const v8h*)(qr + 16 + kh * 8));
            #pragma unroll
            for (int tc = 0; tc < 4; tc++) S[trl][tc] = wmma16(qa, bk[tc], zero8());
        }

        // scale + relative position bias + shift mask, then row softmax
        #pragma unroll
        for (int trl = 0; trl < 2; trl++) {
            int tr = rh * 2 + trl;
            float rmax[8], rsum[8];
            #pragma unroll
            for (int j = 0; j < 8; j++) {
                int rt = tr * 16 + j + 8 * kh;
                int rreg = regionOf(wh * WS + (rt >> 3)) * 3 + regionOf(ww * WS + (rt & 7));
                float mx = -1e30f;
                #pragma unroll
                for (int tc = 0; tc < 4; tc++) {
                    int ct = tc * 16 + ml;
                    int dr = (rt >> 3) - (ct >> 3) + 7;
                    int dc = (rt & 7) - (ct & 7) + 7;
                    float v = S[trl][tc][j] * scale + rpbh[(dr * 15 + dc) * HEADS];
                    if (shift > 0) {
                        int creg = regionOf(wh * WS + (ct >> 3)) * 3 + regionOf(ww * WS + (ct & 7));
                        if (creg != rreg) v -= 100.f;
                    }
                    S[trl][tc][j] = v;
                    mx = fmaxf(mx, v);
                }
                rmax[j] = mx;
            }
            #pragma unroll
            for (int m = 8; m >= 1; m >>= 1)
                #pragma unroll
                for (int j = 0; j < 8; j++) rmax[j] = fmaxf(rmax[j], __shfl_xor(rmax[j], m, 32));
            #pragma unroll
            for (int j = 0; j < 8; j++) {
                float s = 0.f;
                #pragma unroll
                for (int tc = 0; tc < 4; tc++) {
                    float e = __expf(S[trl][tc][j] - rmax[j]);
                    S[trl][tc][j] = e; s += e;
                }
                rsum[j] = s;
            }
            #pragma unroll
            for (int m = 8; m >= 1; m >>= 1)
                #pragma unroll
                for (int j = 0; j < 8; j++) rsum[j] += __shfl_xor(rsum[j], m, 32);
            #pragma unroll
            for (int j = 0; j < 8; j++) {
                float inv = 1.f / rsum[j];
                int rt = tr * 16 + j + 8 * kh;
                #pragma unroll
                for (int tc = 0; tc < 4; tc++)
                    P[rt][tc * 16 + ml] = (_Float16)(S[trl][tc][j] * inv);
            }
        }
    }
    __syncthreads();

    // O(64x32) = P(64x64) . V(64x32); K = 64 -> two chained WMMAs
    v8f O[4][2];
    #pragma unroll
    for (int tr = 0; tr < 4; tr++) { O[tr][0] = zero8(); O[tr][1] = zero8(); }
    #pragma unroll
    for (int kc = 0; kc < 2; kc++) {
        v16h bv[2];
        #pragma unroll
        for (int c = 0; c < 2; c++) {
            const _Float16* vr = qb + 2 * CDIM + c * 16 + ml;
            #pragma unroll
            for (int s = 0; s < 8; s++) {
                bv[c][s]     = vr[(size_t)(kc * 32 + kh * 8 + s) * (3 * CDIM)];
                bv[c][8 + s] = vr[(size_t)(kc * 32 + 16 + kh * 8 + s) * (3 * CDIM)];
            }
        }
        #pragma unroll
        for (int tr = 0; tr < 4; tr++) {
            const _Float16* pr = &P[tr * 16 + ml][kc * 32];
            v16h pa = cat8(*(const v8h*)(pr + kh * 8), *(const v8h*)(pr + 16 + kh * 8));
            O[tr][0] = wmma16(pa, bv[0], O[tr][0]);
            O[tr][1] = wmma16(pa, bv[1], O[tr][1]);
        }
    }
    // write windowed f16 output (Bw, N, C)
    #pragma unroll
    for (int tr = 0; tr < 4; tr++)
        #pragma unroll
        for (int c = 0; c < 2; c++)
            #pragma unroll
            for (int j = 0; j < 8; j++) {
                int row = win * NTOK + tr * 16 + j + 8 * kh;
                int col = head * HD + c * 16 + ml;
                out[(size_t)row * CDIM + col] = (_Float16)O[tr][c][j];
            }
}

// ---------------------------------------------------------------------------
extern "C" void kernel_launch(void* const* d_in, const int* in_sizes, int n_in,
                              void* d_out, int out_size, void* d_ws, size_t ws_size,
                              hipStream_t stream) {
    const float* x       = (const float*)d_in[0];
    const float* norm1_w = (const float*)d_in[1];
    const float* norm1_b = (const float*)d_in[2];
    const float* qkv_w   = (const float*)d_in[3];
    const float* qkv_b   = (const float*)d_in[4];
    const float* rpb     = (const float*)d_in[5];
    const float* proj_w  = (const float*)d_in[6];
    const float* proj_b  = (const float*)d_in[7];
    const float* norm2_w = (const float*)d_in[8];
    const float* norm2_b = (const float*)d_in[9];
    const float* fc1_w   = (const float*)d_in[10];
    const float* fc1_b   = (const float*)d_in[11];
    const float* fc2_w   = (const float*)d_in[12];
    const float* fc2_b   = (const float*)d_in[13];

    char* ws = (char*)d_ws;
    size_t off = 0;
    auto take = [&](size_t bytes) { void* p = ws + off; off += (bytes + 255) & ~(size_t)255; return p; };

    _Float16* wqkv_h = (_Float16*)take((size_t)2 * CDIM * 3 * CDIM * 2);
    _Float16* wprj_h = (_Float16*)take((size_t)2 * CDIM * CDIM * 2);
    _Float16* wfc1_h = (_Float16*)take((size_t)2 * CDIM * HID * 2);
    _Float16* wfc2_h = (_Float16*)take((size_t)2 * HID * CDIM * 2);
    _Float16* xh     = (_Float16*)take((size_t)MROWS * CDIM * 2);   // windowed LN (reused for LN2)
    _Float16* qkvb   = (_Float16*)take((size_t)MROWS * 3 * CDIM * 2);
    _Float16* attn_o = (_Float16*)take((size_t)MROWS * CDIM * 2);
    float*    y      = (float*)   take((size_t)MROWS * CDIM * 4);
    _Float16* h1     = (_Float16*)take((size_t)MROWS * HID * 2);
    float*    x1     = (float*)   take((size_t)MROWS * CDIM * 4);

    auto cvt = [&](const float* s, _Float16* d, int n) {
        cvt_f32_f16_kernel<<<(n + 255) / 256, 256, 0, stream>>>(s, d, n);
    };
    cvt(qkv_w,  wqkv_h, 2 * CDIM * 3 * CDIM);
    cvt(proj_w, wprj_h, 2 * CDIM * CDIM);
    cvt(fc1_w,  wfc1_h, 2 * CDIM * HID);
    cvt(fc2_w,  wfc2_h, 2 * HID * CDIM);

    for (int i = 0; i < 2; i++) {
        const int shift = (i % 2 == 0) ? 0 : WS / 2;
        const float* xin  = (i == 0) ? x : x1;
        float*       xout = (i == 1) ? (float*)d_out : x1;

        // LN1 + shift-roll + window partition -> f16
        ln_kernel<<<MROWS / 4, 128, 0, stream>>>(xin, norm1_w + i * CDIM, norm1_b + i * CDIM,
                                                 xh, shift, 1);
        // QKV GEMM (mode 0)
        gemm_wmma_kernel<<<dim3((3 * CDIM) / 128, MROWS / 256), 256, 0, stream>>>(
            xh, wqkv_h + (size_t)i * CDIM * 3 * CDIM, qkv_b + i * 3 * CDIM,
            MROWS, 3 * CDIM, CDIM, qkvb, nullptr, nullptr, 0, 0);
        // fused window attention
        attn_kernel<<<BW * HEADS, 32, 0, stream>>>(qkvb, rpb + i * 225 * HEADS, attn_o, shift);
        // proj GEMM + window reverse + residual (mode 3) -> y (f32, image layout)
        gemm_wmma_kernel<<<dim3(CDIM / 128, MROWS / 256), 256, 0, stream>>>(
            attn_o, wprj_h + (size_t)i * CDIM * CDIM, proj_b + i * CDIM,
            MROWS, CDIM, CDIM, nullptr, y, xin, 3, shift);
        // LN2 (plain layout) -> f16
        ln_kernel<<<MROWS / 4, 128, 0, stream>>>(y, norm2_w + i * CDIM, norm2_b + i * CDIM,
                                                 xh, 0, 0);
        // fc1 GEMM + GELU (mode 1)
        gemm_wmma_kernel<<<dim3(HID / 128, MROWS / 256), 256, 0, stream>>>(
            xh, wfc1_h + (size_t)i * CDIM * HID, fc1_b + i * HID,
            MROWS, HID, CDIM, h1, nullptr, nullptr, 1, 0);
        // fc2 GEMM + residual (mode 2) -> block output (f32)
        gemm_wmma_kernel<<<dim3(CDIM / 128, MROWS / 256), 256, 0, stream>>>(
            h1, wfc2_h + (size_t)i * HID * CDIM, fc2_b + i * CDIM,
            MROWS, CDIM, HID, nullptr, xout, y, 2, 0);
    }
    (void)in_sizes; (void)n_in; (void)out_size; (void)ws_size;
}
